// Pcm_618475291307
// MI455X (gfx1250) — matrix operand order repaired and checked by reading, not compiled, not run on hardware
//
#include <hip/hip_runtime.h>
#include <cstdint>
#include <cstddef>

// MI455X / gfx1250, wave32. f16 WMMA (v_wmma_f32_16x16x32_f16) for both
// q·qT and attn·V. Workspace: Q in f16, 128 windows * 1024 * 64 * 2B = 16 MB.

typedef _Float16 v16h __attribute__((ext_vector_type(16)));
typedef _Float16 v8h  __attribute__((ext_vector_type(8)));
typedef _Float16 h2   __attribute__((ext_vector_type(2)));
typedef float    v8f  __attribute__((ext_vector_type(8)));

#define NCH   128
#define CDIM  64
#define NPIX  1024
#define HW    256
#define PLANE 65536   // 256*256

// ---- LDS fragment loaders (per ISA 7.12.2 wave32 layouts) ----
// A (16x32 f16): lane L -> row = L%16; vgpr r -> K = (r<4?0:16) + (L/16)*8 + 2*(r%4)
__device__ inline v16h ldsA(const _Float16* base, int rowBase, int kOff, int lane) {
  v16h a;
  const int row = (rowBase + (lane & 15)) * 64;
  const int hi  = (lane >> 4) * 8;
  #pragma unroll
  for (int r = 0; r < 8; ++r) {
    const int K = ((r & 4) << 2) + hi + ((r & 3) << 1);
    h2 p = *(const h2*)(base + row + kOff + K);
    a[2*r] = p[0]; a[2*r+1] = p[1];
  }
  return a;
}
// B (32x16 f16): lane L -> col n = L%16; vgpr r -> K = (r<4?0:16) + (L/16)*8 + 2*(r%4)
// source matrix stored row-major with row index = n, column offset = K.
__device__ inline v16h ldsB(const _Float16* base, int nBase, int kOff, int lane) {
  v16h b;
  const int row = (nBase + (lane & 15)) * 64;
  const int hi  = (lane >> 4) * 8;
  #pragma unroll
  for (int r = 0; r < 8; ++r) {
    const int K = ((r & 4) << 2) + hi + ((r & 3) << 1);
    h2 p = *(const h2*)(base + row + kOff + K);
    b[2*r] = p[0]; b[2*r+1] = p[1];
  }
  return b;
}

// ================= Kernel 1: q = normalize(W_qk @ x_window), f16 to ws ======
__global__ __launch_bounds__(256) void q_kernel(const float* __restrict__ x,
                                                const float* __restrict__ Wqk,
                                                _Float16* __restrict__ qws) {
  __shared__ float sW[NCH * CDIM];          // W transposed: sW[n][c], 32 KB
  const int t = threadIdx.x;
  const int win = blockIdx.y;               // 0..127 : (bb, n1i, n2i)
  const int chunk = blockIdx.x;             // 0..3   : 256-pixel chunk
  const int bb = win >> 6, n1i = (win >> 3) & 7, n2i = win & 7;

  for (int e = t; e < NCH * CDIM; e += 256) {
    const int c = e >> 7, n = e & 127;
    sW[n * CDIM + c] = Wqk[e];
  }
  __syncthreads();

  const int pix = chunk * 256 + t;          // 0..1023 within window
  const int p1 = pix >> 5, p2 = pix & 31;
  const float* xp = x + (size_t)bb * NCH * PLANE
                      + (size_t)(n1i * 32 + p1) * HW + n2i * 32 + p2;

  float acc[CDIM];
  #pragma unroll
  for (int c = 0; c < CDIM; ++c) acc[c] = 0.0f;

  for (int n = 0; n < NCH; ++n) {
    const float xv = xp[(size_t)n * PLANE];
    const float4* w4 = (const float4*)(sW + n * CDIM);
    #pragma unroll
    for (int k = 0; k < 16; ++k) {
      const float4 wv = w4[k];
      acc[4*k+0] += wv.x * xv;
      acc[4*k+1] += wv.y * xv;
      acc[4*k+2] += wv.z * xv;
      acc[4*k+3] += wv.w * xv;
    }
  }
  float ss = 0.0f;
  #pragma unroll
  for (int c = 0; c < CDIM; ++c) ss += acc[c] * acc[c];
  const float inv = 1.0f / (sqrtf(ss) + 1e-6f);

  v8h* dst = (v8h*)(qws + ((size_t)win * NPIX + pix) * CDIM);
  #pragma unroll
  for (int g = 0; g < 8; ++g) {
    v8h hv;
    #pragma unroll
    for (int k2 = 0; k2 < 8; ++k2) hv[k2] = (_Float16)(acc[g*8 + k2] * inv);
    dst[g] = hv;
  }
}

// ================= Kernel 2: streamed relu-attention via WMMA ===============
// grid: (4 i-tiles, 128 windows), 256 threads = 8 waves.
// LDS: sQ 128KB | sS 32KB | sV 16KB | sRow 1KB  => 181,248 B (dynamic)
__global__ __launch_bounds__(256) void attn_kernel(const float* __restrict__ vg,
                                                   const _Float16* __restrict__ qws,
                                                   float* __restrict__ out) {
  extern __shared__ char smem[];
  _Float16* sQ   = (_Float16*)smem;          // [1024][64]
  _Float16* sS   = sQ + NPIX * CDIM;         // [256][64]  relu(S) f16, j-tile
  _Float16* sV   = sS + 256 * 64;            // [128][64]  V tile, [ch][j]
  float*    sRow = (float*)(sV + NCH * 64);  // [256] row sums

  const int t = threadIdx.x;
  const int lane = t & 31;
  const int w = t >> 5;                       // wave 0..7
  const int win = blockIdx.y;
  const int i0 = blockIdx.x * 256;            // i-tile base (window-local)
  const int bb = win >> 6, n1i = (win >> 3) & 7, n2i = win & 7;

  sRow[t] = 0.0f;

  { // stage whole-window Q (128 KB) into LDS
    const uint4* src = (const uint4*)(qws + (size_t)win * NPIX * CDIM);
    uint4* dst = (uint4*)sQ;
    #pragma unroll
    for (int k = 0; k < 32; ++k) dst[t + k * 256] = src[t + k * 256];
  }
  __syncthreads();

  // persistent A fragments: this wave's 32 i-rows, K = c (two 32-wide halves)
  v16h aQ[2][2];
  #pragma unroll
  for (int m = 0; m < 2; ++m) {
    aQ[m][0] = ldsA(sQ, i0 + (2*w + m) * 16, 0,  lane);
    aQ[m][1] = ldsA(sQ, i0 + (2*w + m) * 16, 32, lane);
  }

  v8f acc[2][8];
  #pragma unroll
  for (int m = 0; m < 2; ++m)
    #pragma unroll
    for (int n = 0; n < 8; ++n) acc[m][n] = {};

  const float* vwin = vg + (size_t)bb * NCH * PLANE
                         + (size_t)(n1i * 32) * HW + n2i * 32;

  for (int jj = 0; jj < 16; ++jj) {
    const int j0 = jj * 64;

    // ---- stage V tile: 64 j-pixels x 128 ch -> sV[ch][j] as f16 ----
    #pragma unroll
    for (int it = 0; it < 16; ++it) {
      const int pe = t + it * 256;            // pair index 0..4095
      const int ch = pe >> 5;
      const int jl = (pe & 31) << 1;          // even j within tile
      const int jpix = j0 + jl;
      const int p1 = jpix >> 5, p2 = jpix & 31;
      const float2 xv = *(const float2*)(vwin + (size_t)ch * PLANE + p1 * HW + p2);
      h2 hv; hv[0] = (_Float16)xv.x; hv[1] = (_Float16)xv.y;
      *(h2*)(sV + ch * 64 + jl) = hv;
    }

    // ---- S = relu(Q_i . Q_jT), rows owned by this wave ----
    #pragma unroll
    for (int jt = 0; jt < 4; ++jt) {
      const v16h b0 = ldsB(sQ, j0 + jt * 16, 0,  lane);
      const v16h b1 = ldsB(sQ, j0 + jt * 16, 32, lane);
      #pragma unroll
      for (int m = 0; m < 2; ++m) {
        v8f d = {};
        d = __builtin_amdgcn_wmma_f32_16x16x32_f16(false, aQ[m][0], false, b0,
                                                   (short)0, d, false, false);
        d = __builtin_amdgcn_wmma_f32_16x16x32_f16(false, aQ[m][1], false, b1,
                                                   (short)0, d, false, false);
        const int rbase = (2*w + m) * 16;
        #pragma unroll
        for (int e = 0; e < 8; ++e) {
          const float val = d[e] > 0.0f ? d[e] : 0.0f;
          const int row = rbase + e + ((lane >> 4) << 3);
          sS[row * 64 + jt * 16 + (lane & 15)] = (_Float16)val;
          float s = val;                       // 16-lane group reduction
          s += __shfl_xor(s, 1, 32);
          s += __shfl_xor(s, 2, 32);
          s += __shfl_xor(s, 4, 32);
          s += __shfl_xor(s, 8, 32);
          if ((lane & 15) == 0) atomicAdd(&sRow[row], s);
        }
      }
    }
    __syncthreads();

    // ---- acc += S (32 x 64) x V (64 x 128) ----
    v16h aS[2][2];
    #pragma unroll
    for (int m = 0; m < 2; ++m) {
      aS[m][0] = ldsA(sS, (2*w + m) * 16, 0,  lane);
      aS[m][1] = ldsA(sS, (2*w + m) * 16, 32, lane);
    }
    #pragma unroll
    for (int nt = 0; nt < 8; ++nt) {
      const v16h bv0 = ldsB(sV, nt * 16, 0,  lane);
      const v16h bv1 = ldsB(sV, nt * 16, 32, lane);
      #pragma unroll
      for (int m = 0; m < 2; ++m) {
        acc[m][nt] = __builtin_amdgcn_wmma_f32_16x16x32_f16(false, aS[m][0], false, bv0,
                                                            (short)0, acc[m][nt], false, false);
        acc[m][nt] = __builtin_amdgcn_wmma_f32_16x16x32_f16(false, aS[m][1], false, bv1,
                                                            (short)0, acc[m][nt], false, false);
      }
    }
    __syncthreads();
  }

  // ---- finalize: scale by 1/(rowsum+1e-6), transpose via LDS, store ----
  float* stg = (float*)smem + w * (NCH * 33);      // wave-private, stride-33 pad
  const int p1o = (i0 + w * 32) >> 5;              // window-local p1 of this wave
  float* obase = out + (size_t)bb * NCH * PLANE
                     + (size_t)(n1i * 32 + p1o) * HW + n2i * 32;

  #pragma unroll
  for (int m = 0; m < 2; ++m) {
    #pragma unroll
    for (int e = 0; e < 8; ++e) {
      const int row = (2*w + m) * 16 + e + ((lane >> 4) << 3);
      const float sc = 1.0f / (sRow[row] + 1e-6f);
      const int r32 = row - w * 32;                // 0..31 == p2
      #pragma unroll
      for (int nt = 0; nt < 8; ++nt) {
        const int ch = nt * 16 + (lane & 15);
        stg[ch * 33 + r32] = acc[m][nt][e] * sc;
      }
    }
  }
  __syncthreads();
  #pragma unroll 4
  for (int ch = 0; ch < NCH; ++ch) {
    obase[(size_t)ch * PLANE + lane] = stg[ch * 33 + lane];  // 128B coalesced
  }
}

// ============================================================================
extern "C" void kernel_launch(void* const* d_in, const int* in_sizes, int n_in,
                              void* d_out, int out_size, void* d_ws, size_t ws_size,
                              hipStream_t stream) {
  const float* x = (const float*)d_in[0];
  const float* v = (const float*)d_in[1];
  const float* W = (const float*)d_in[2];
  float* out = (float*)d_out;
  _Float16* qws = (_Float16*)d_ws;   // needs 16 MB: 128 * 1024 * 64 * 2B

  dim3 blk(256);
  q_kernel<<<dim3(4, 128), blk, 0, stream>>>(x, W, qws);

  const size_t lds = (size_t)(NPIX * CDIM + 256 * 64 + NCH * 64) * sizeof(_Float16)
                   + 256 * sizeof(float);          // 181,248 B (< 320 KB/WGP)
  attn_kernel<<<dim3(4, 128), blk, lds, stream>>>(v, qws, out);
}
